// OptimizedMoELayer_79534204387708
// MI455X (gfx1250) — compile-verified, matrix operand order checked by reading, not run on hardware
//
#include <hip/hip_runtime.h>

// ---------------- problem constants (from reference) ----------------
constexpr int T_  = 16384;   // tokens
constexpr int H_  = 1024;    // hidden
constexpr int F_  = 4096;    // ffn
constexpr int E_  = 8;       // experts
constexpr int C_  = 2560;    // capacity = ceil(1.25*T/E)
constexpr int NB_ = 64;      // histogram blocks (256 tokens each)

// ---------------- vector types for WMMA ----------------
typedef __attribute__((ext_vector_type(16))) __bf16 v16bf;
typedef __attribute__((ext_vector_type(8)))  __bf16 v8bf;
typedef __attribute__((ext_vector_type(4)))  __bf16 v4bf;
typedef __attribute__((ext_vector_type(8)))  float  v8f;
typedef int v4i_ __attribute__((vector_size(16)));   // matches builtin param type

__device__ __forceinline__ __bf16 f2bf(float f) {
    unsigned u = __builtin_bit_cast(unsigned, f);
    unsigned r = u + 0x7FFFu + ((u >> 16) & 1u);   // round-to-nearest-even
    unsigned short s = (unsigned short)(r >> 16);
    return __builtin_bit_cast(__bf16, s);
}

// ---------------- async global->LDS copy (CDNA5 ASYNCcnt path) ----------------
#if __has_builtin(__builtin_amdgcn_global_load_async_to_lds_b128)
#define HAVE_ASYNC_LDS 1
__device__ __forceinline__ void copy16_g2l(__bf16* lds_dst, const __bf16* gsrc) {
    __builtin_amdgcn_global_load_async_to_lds_b128(
        (__attribute__((address_space(1))) v4i_*)gsrc,
        (__attribute__((address_space(3))) v4i_*)lds_dst, 0, 0);
}
__device__ __forceinline__ void async_wait() {
#if __has_builtin(__builtin_amdgcn_s_wait_asynccnt)
    __builtin_amdgcn_s_wait_asynccnt(0);
#else
    asm volatile("s_wait_asynccnt 0x0" ::: "memory");
#endif
}
#else
__device__ __forceinline__ void copy16_g2l(__bf16* lds_dst, const __bf16* gsrc) {
    *(v8bf*)lds_dst = *(const v8bf*)gsrc;
}
__device__ __forceinline__ void async_wait() {}
#endif

// =====================================================================
// 1) Router: logits, softmax, top-2, normalized weights. One wave/token.
// =====================================================================
__global__ __launch_bounds__(256) void moe_router(
    const float* __restrict__ tokens, const float* __restrict__ wr,
    int* __restrict__ topi, float* __restrict__ topw)
{
    int lane = threadIdx.x & 31, wid = threadIdx.x >> 5;
    int t = blockIdx.x * 8 + wid;
    float acc[E_];
#pragma unroll
    for (int e = 0; e < E_; ++e) acc[e] = 0.f;
    const float* tok = tokens + (size_t)t * H_;
    for (int h = lane; h < H_; h += 32) {
        float x = tok[h];
        const float* w = wr + (size_t)h * E_;
#pragma unroll
        for (int e = 0; e < E_; ++e) acc[e] += x * w[e];
    }
#pragma unroll
    for (int off = 16; off >= 1; off >>= 1) {
#pragma unroll
        for (int e = 0; e < E_; ++e) acc[e] += __shfl_xor(acc[e], off, 32);
    }
    if (lane == 0) {
        float m = acc[0];
#pragma unroll
        for (int e = 1; e < E_; ++e) m = acc[e] > m ? acc[e] : m;
        float p[E_];
#pragma unroll
        for (int e = 0; e < E_; ++e) p[e] = __expf(acc[e] - m);
        int i0 = 0; float v0 = p[0];
#pragma unroll
        for (int e = 1; e < E_; ++e) if (p[e] > v0) { v0 = p[e]; i0 = e; }
        int i1 = -1; float v1 = -1.f;
#pragma unroll
        for (int e = 0; e < E_; ++e) if (e != i0 && p[e] > v1) { v1 = p[e]; i1 = e; }
        float s = v0 + v1;
        topi[t * 2 + 0] = i0;  topi[t * 2 + 1] = i1;
        topw[t * 2 + 0] = v0 / s;  topw[t * 2 + 1] = v1 / s;
    }
}

// =====================================================================
// 2) Per-block (k,e) histogram over 256-token chunks
// =====================================================================
__global__ __launch_bounds__(256) void moe_hist(
    const int* __restrict__ topi, int* __restrict__ blockcnt)
{
    __shared__ int cnt[16];
    if (threadIdx.x < 16) cnt[threadIdx.x] = 0;
    __syncthreads();
    int t = blockIdx.x * 256 + threadIdx.x;
    atomicAdd(&cnt[0 + topi[t * 2 + 0]], 1);
    atomicAdd(&cnt[8 + topi[t * 2 + 1]], 1);
    __syncthreads();
    if (threadIdx.x < 16) blockcnt[blockIdx.x * 16 + threadIdx.x] = cnt[threadIdx.x];
}

// =====================================================================
// 3) Exclusive scan over blocks + slot_off (k=1 starts after all k=0)
// =====================================================================
__global__ void moe_scan(const int* __restrict__ blockcnt, int* __restrict__ blockbase)
{
    __shared__ int tot[16];
    int i = threadIdx.x;
    if (i < 16) {
        int run = 0;
        for (int b = 0; b < NB_; ++b) {
            blockbase[b * 16 + i] = run;
            run += blockcnt[b * 16 + i];
        }
        tot[i] = run;
    }
    __syncthreads();
    if (i < 16) {
        int off = (i >= 8) ? tot[i - 8] : 0;   // second choice after all first-choice slots
        for (int b = 0; b < NB_; ++b) blockbase[b * 16 + i] += off;
    }
}

// =====================================================================
// 4) In-token-order rank assignment within each 256-token chunk
// =====================================================================
__global__ __launch_bounds__(256) void moe_rank(
    const int* __restrict__ topi, const int* __restrict__ blockbase,
    int* __restrict__ slot)
{
    __shared__ int le[2][256];
    __shared__ int ls[2][256];
    int t = blockIdx.x * 256 + threadIdx.x;
    le[0][threadIdx.x] = topi[t * 2 + 0];
    le[1][threadIdx.x] = topi[t * 2 + 1];
    __syncthreads();
    if (threadIdx.x < 16) {
        int k = threadIdx.x >> 3, e = threadIdx.x & 7;
        int c = blockbase[blockIdx.x * 16 + threadIdx.x];
        for (int j = 0; j < 256; ++j)
            if (le[k][j] == e) ls[k][j] = c++;
    }
    __syncthreads();
    slot[t * 2 + 0] = ls[0][threadIdx.x];
    slot[t * 2 + 1] = ls[1][threadIdx.x];
}

// =====================================================================
// 5) Zero Xbuf (16B per thread)
// =====================================================================
__global__ __launch_bounds__(256) void moe_zero16(uint4* __restrict__ p)
{
    size_t i = (size_t)blockIdx.x * 256 + threadIdx.x;
    p[i] = make_uint4(0u, 0u, 0u, 0u);
}

// =====================================================================
// 6) Gather tokens into expert buffers (fp32 -> bf16). One wave/(t,k).
// =====================================================================
__global__ __launch_bounds__(256) void moe_gather(
    const float* __restrict__ tokens, const int* __restrict__ topi,
    const int* __restrict__ slot, __bf16* __restrict__ Xbuf)
{
    int lane = threadIdx.x & 31, wid = threadIdx.x >> 5;
    int p = blockIdx.x * 8 + wid;        // (token,k) pair index
    int r = slot[p];
    if (r >= C_) return;                 // dropped
    int e = topi[p];
    int t = p >> 1;
    const float* src = tokens + (size_t)t * H_;
    __bf16* dst = Xbuf + ((size_t)e * C_ + r) * H_;
#pragma unroll
    for (int i = 0; i < 8; ++i) {
        int h = i * 128 + lane * 4;
        float4 v = *(const float4*)(src + h);
        v4bf b; b[0] = f2bf(v.x); b[1] = f2bf(v.y); b[2] = f2bf(v.z); b[3] = f2bf(v.w);
        *(v4bf*)(dst + h) = b;
    }
}

// =====================================================================
// 7) Fused gate/up GEMM + SiLU: H = silu(X@Wg) * (X@Wu), bf16 out.
//    128x64 block tile, BK=32, double-buffered LDS, async A staging.
// =====================================================================
#define LDSTR 40   // 32 + 8 pad halves -> conflict-free ds_load_b128 across rows

__global__ __launch_bounds__(256) void moe_gemm_gateup(
    const __bf16* __restrict__ Xbuf, const float* __restrict__ Wg,
    const float* __restrict__ Wu, __bf16* __restrict__ Hbuf)
{
    int e = blockIdx.z;
    int bm0 = blockIdx.y * 128;
    int bn0 = blockIdx.x * 64;
    const __bf16* X = Xbuf + (size_t)e * C_ * H_;
    const float*  G = Wg   + (size_t)e * H_ * F_;
    const float*  U = Wu   + (size_t)e * H_ * F_;
    __bf16*       Hb = Hbuf + (size_t)e * C_ * F_;

    __shared__ __bf16 As[2][128 * LDSTR];
    __shared__ __bf16 Bg[2][64 * LDSTR];
    __shared__ __bf16 Bu[2][64 * LDSTR];

    int tid = threadIdx.x, lane = tid & 31, wid = tid >> 5;
    int wm = (wid & 3) * 32;   // 4 waves along M
    int wn = (wid >> 2) * 32;  // 2 waves along N

    v8f accg[2][2], accu[2][2];
#pragma unroll
    for (int mi = 0; mi < 2; ++mi)
#pragma unroll
        for (int ni = 0; ni < 2; ++ni) { accg[mi][ni] = (v8f)0.f; accu[mi][ni] = (v8f)0.f; }

    constexpr int NT = H_ / 32;
    float4 vg[2], vu[2];

    // A tile: 128x32 bf16 = 512 16B chunks, async copy (no VGPR transit)
    auto stageA = [&](int kt, int b) {
#pragma unroll
        for (int i = 0; i < 2; ++i) {
            int c = tid + i * 256;
            int row = c >> 2, cg = (c & 3) * 8;
            copy16_g2l(&As[b][row * LDSTR + cg],
                       X + (size_t)(bm0 + row) * H_ + kt * 32 + cg);
        }
    };
    // B tiles: issue global loads (fp32) into registers
    auto loadB = [&](int kt) {
#pragma unroll
        for (int r = 0; r < 2; ++r) {
            int g = tid + r * 256;
            int k = g >> 4, ng = (g & 15) * 4;
            vg[r] = *(const float4*)(G + (size_t)(kt * 32 + k) * F_ + bn0 + ng);
            vu[r] = *(const float4*)(U + (size_t)(kt * 32 + k) * F_ + bn0 + ng);
        }
    };
    // convert + transpose-store B tiles into LDS [n][k]
    auto storeB = [&](int b) {
#pragma unroll
        for (int r = 0; r < 2; ++r) {
            int g = tid + r * 256;
            int k = g >> 4, ng = (g & 15) * 4;
            Bg[b][(ng + 0) * LDSTR + k] = f2bf(vg[r].x);
            Bg[b][(ng + 1) * LDSTR + k] = f2bf(vg[r].y);
            Bg[b][(ng + 2) * LDSTR + k] = f2bf(vg[r].z);
            Bg[b][(ng + 3) * LDSTR + k] = f2bf(vg[r].w);
            Bu[b][(ng + 0) * LDSTR + k] = f2bf(vu[r].x);
            Bu[b][(ng + 1) * LDSTR + k] = f2bf(vu[r].y);
            Bu[b][(ng + 2) * LDSTR + k] = f2bf(vu[r].z);
            Bu[b][(ng + 3) * LDSTR + k] = f2bf(vu[r].w);
        }
    };

    loadB(0);
    stageA(0, 0);
    storeB(0);
    async_wait();
    __syncthreads();

    for (int kt = 0; kt < NT; ++kt) {
        int cur = kt & 1, nxt = cur ^ 1;
        if (kt + 1 < NT) {           // issue next tile's loads before computing
            loadB(kt + 1);
            stageA(kt + 1, nxt);
        }
        if (kt + 2 < NT) {           // prefetch B two tiles ahead
            int g = tid;
            int k = g >> 4, ng = (g & 15) * 4;
            __builtin_prefetch(G + (size_t)((kt + 2) * 32 + k) * F_ + bn0 + ng, 0, 1);
            __builtin_prefetch(U + (size_t)((kt + 2) * 32 + k) * F_ + bn0 + ng, 0, 1);
        }

        // compute current tile
        {
            int lr = lane & 15, lk = (lane >> 4) * 8;
            v16bf af[2], bgf[2], buf2[2];
#pragma unroll
            for (int mi = 0; mi < 2; ++mi) {
                const __bf16* p = &As[cur][(wm + mi * 16 + lr) * LDSTR + lk];
                v8bf lo = *(const v8bf*)p, hi = *(const v8bf*)(p + 16);
#pragma unroll
                for (int i = 0; i < 8; ++i) { af[mi][i] = lo[i]; af[mi][i + 8] = hi[i]; }
            }
#pragma unroll
            for (int ni = 0; ni < 2; ++ni) {
                const __bf16* pg = &Bg[cur][(wn + ni * 16 + lr) * LDSTR + lk];
                const __bf16* pu = &Bu[cur][(wn + ni * 16 + lr) * LDSTR + lk];
                v8bf glo = *(const v8bf*)pg, ghi = *(const v8bf*)(pg + 16);
                v8bf ulo = *(const v8bf*)pu, uhi = *(const v8bf*)(pu + 16);
#pragma unroll
                for (int i = 0; i < 8; ++i) {
                    bgf[ni][i] = glo[i]; bgf[ni][i + 8] = ghi[i];
                    buf2[ni][i] = ulo[i]; buf2[ni][i + 8] = uhi[i];
                }
            }
#pragma unroll
            for (int mi = 0; mi < 2; ++mi)
#pragma unroll
                for (int ni = 0; ni < 2; ++ni) {
                    accg[mi][ni] = __builtin_amdgcn_wmma_f32_16x16x32_bf16(
                        false, af[mi], false, bgf[ni], (short)0, accg[mi][ni], false, false);
                    accu[mi][ni] = __builtin_amdgcn_wmma_f32_16x16x32_bf16(
                        false, af[mi], false, buf2[ni], (short)0, accu[mi][ni], false, false);
                }
        }

        if (kt + 1 < NT) storeB(nxt);   // convert+store after WMMAs (loads landed)
        async_wait();
        __syncthreads();
    }

    // epilogue: silu(g)*u -> bf16 (rcp instead of full-precision divide)
    int lr = lane & 15, lhi = (lane >> 4) * 8;
#pragma unroll
    for (int mi = 0; mi < 2; ++mi)
#pragma unroll
        for (int ni = 0; ni < 2; ++ni)
#pragma unroll
            for (int vi = 0; vi < 8; ++vi) {
                int row = bm0 + wm + mi * 16 + lhi + vi;
                int col = bn0 + wn + ni * 16 + lr;
                float gv = accg[mi][ni][vi], uv = accu[mi][ni][vi];
                float hv = gv * __builtin_amdgcn_rcpf(1.f + __expf(-gv)) * uv;
                Hb[(size_t)row * F_ + col] = f2bf(hv);
            }
}

// =====================================================================
// 8) Down GEMM: Y = H @ Wd, fp32 out. 128x128 tile, double-buffered.
// =====================================================================
__global__ __launch_bounds__(256) void moe_gemm_down(
    const __bf16* __restrict__ Hbuf, const float* __restrict__ Wd,
    float* __restrict__ Ybuf)
{
    int e = blockIdx.z;
    int bm0 = blockIdx.y * 128;
    int bn0 = blockIdx.x * 128;
    const __bf16* Hb = Hbuf + (size_t)e * C_ * F_;
    const float*  W  = Wd   + (size_t)e * F_ * H_;
    float*        Y  = Ybuf + (size_t)e * C_ * H_;

    __shared__ __bf16 As[2][128 * LDSTR];
    __shared__ __bf16 Bs[2][128 * LDSTR];

    int tid = threadIdx.x, lane = tid & 31, wid = tid >> 5;
    int wm = (wid & 3) * 32;
    int wn = (wid >> 2) * 64;

    v8f acc[2][4];
#pragma unroll
    for (int mi = 0; mi < 2; ++mi)
#pragma unroll
        for (int ni = 0; ni < 4; ++ni) acc[mi][ni] = (v8f)0.f;

    constexpr int NT = F_ / 32;
    float4 vb[4];

    auto stageA = [&](int kt, int b) {
#pragma unroll
        for (int i = 0; i < 2; ++i) {
            int c = tid + i * 256;
            int row = c >> 2, cg = (c & 3) * 8;
            copy16_g2l(&As[b][row * LDSTR + cg],
                       Hb + (size_t)(bm0 + row) * F_ + kt * 32 + cg);
        }
    };
    auto loadB = [&](int kt) {
#pragma unroll
        for (int r = 0; r < 4; ++r) {
            int g = tid + r * 256;
            int k = g >> 5, ng = (g & 31) * 4;
            vb[r] = *(const float4*)(W + (size_t)(kt * 32 + k) * H_ + bn0 + ng);
        }
    };
    auto storeB = [&](int b) {
#pragma unroll
        for (int r = 0; r < 4; ++r) {
            int g = tid + r * 256;
            int k = g >> 5, ng = (g & 31) * 4;
            Bs[b][(ng + 0) * LDSTR + k] = f2bf(vb[r].x);
            Bs[b][(ng + 1) * LDSTR + k] = f2bf(vb[r].y);
            Bs[b][(ng + 2) * LDSTR + k] = f2bf(vb[r].z);
            Bs[b][(ng + 3) * LDSTR + k] = f2bf(vb[r].w);
        }
    };

    loadB(0);
    stageA(0, 0);
    storeB(0);
    async_wait();
    __syncthreads();

    for (int kt = 0; kt < NT; ++kt) {
        int cur = kt & 1, nxt = cur ^ 1;
        if (kt + 1 < NT) {
            loadB(kt + 1);
            stageA(kt + 1, nxt);
        }
        if (kt + 2 < NT) {
            int g = tid;
            int k = g >> 5, ng = (g & 31) * 4;
            __builtin_prefetch(W + (size_t)((kt + 2) * 32 + k) * H_ + bn0 + ng, 0, 1);
        }

        {
            int lr = lane & 15, lk = (lane >> 4) * 8;
            v16bf af[2], bf[4];
#pragma unroll
            for (int mi = 0; mi < 2; ++mi) {
                const __bf16* p = &As[cur][(wm + mi * 16 + lr) * LDSTR + lk];
                v8bf lo = *(const v8bf*)p, hi = *(const v8bf*)(p + 16);
#pragma unroll
                for (int i = 0; i < 8; ++i) { af[mi][i] = lo[i]; af[mi][i + 8] = hi[i]; }
            }
#pragma unroll
            for (int ni = 0; ni < 4; ++ni) {
                const __bf16* p = &Bs[cur][(wn + ni * 16 + lr) * LDSTR + lk];
                v8bf lo = *(const v8bf*)p, hi = *(const v8bf*)(p + 16);
#pragma unroll
                for (int i = 0; i < 8; ++i) { bf[ni][i] = lo[i]; bf[ni][i + 8] = hi[i]; }
            }
#pragma unroll
            for (int mi = 0; mi < 2; ++mi)
#pragma unroll
                for (int ni = 0; ni < 4; ++ni)
                    acc[mi][ni] = __builtin_amdgcn_wmma_f32_16x16x32_bf16(
                        false, af[mi], false, bf[ni], (short)0, acc[mi][ni], false, false);
        }

        if (kt + 1 < NT) storeB(nxt);
        async_wait();
        __syncthreads();
    }

    int lr = lane & 15, lhi = (lane >> 4) * 8;
#pragma unroll
    for (int mi = 0; mi < 2; ++mi)
#pragma unroll
        for (int ni = 0; ni < 4; ++ni)
#pragma unroll
            for (int vi = 0; vi < 8; ++vi) {
                int row = bm0 + wm + mi * 16 + lhi + vi;
                int col = bn0 + wn + ni * 16 + lr;
                Y[(size_t)row * H_ + col] = acc[mi][ni][vi];
            }
}

// =====================================================================
// 9) Weighted scatter back to output. One wave/token.
// =====================================================================
__global__ __launch_bounds__(256) void moe_scatter(
    const float* __restrict__ Ybuf, const int* __restrict__ topi,
    const int* __restrict__ slot, const float* __restrict__ topw,
    float* __restrict__ out)
{
    int lane = threadIdx.x & 31, wid = threadIdx.x >> 5;
    int t = blockIdx.x * 8 + wid;
    int e0 = topi[t * 2 + 0], e1 = topi[t * 2 + 1];
    int r0 = slot[t * 2 + 0], r1 = slot[t * 2 + 1];
    float w0 = (r0 < C_) ? topw[t * 2 + 0] : 0.f;
    float w1 = (r1 < C_) ? topw[t * 2 + 1] : 0.f;
    int c0 = r0 < C_ ? r0 : C_ - 1;
    int c1 = r1 < C_ ? r1 : C_ - 1;
    const float* y0 = Ybuf + ((size_t)e0 * C_ + c0) * H_;
    const float* y1 = Ybuf + ((size_t)e1 * C_ + c1) * H_;
    float* dst = out + (size_t)t * H_;
#pragma unroll
    for (int i = 0; i < 8; ++i) {
        int h = i * 128 + lane * 4;
        float4 a = *(const float4*)(y0 + h);
        float4 b = *(const float4*)(y1 + h);
        float4 o;
        o.x = w0 * a.x + w1 * b.x;
        o.y = w0 * a.y + w1 * b.y;
        o.z = w0 * a.z + w1 * b.z;
        o.w = w0 * a.w + w1 * b.w;
        *(float4*)(dst + h) = o;
    }
}

// =====================================================================
// launch
// =====================================================================
extern "C" void kernel_launch(void* const* d_in, const int* in_sizes, int n_in,
                              void* d_out, int out_size, void* d_ws, size_t ws_size,
                              hipStream_t stream) {
    (void)in_sizes; (void)n_in; (void)out_size; (void)ws_size;
    const float* tokens   = (const float*)d_in[0];
    const float* w_router = (const float*)d_in[1];
    const float* w_gate   = (const float*)d_in[2];
    const float* w_up     = (const float*)d_in[3];
    const float* w_down   = (const float*)d_in[4];
    float* out = (float*)d_out;

    char* ws = (char*)d_ws;
    size_t off = 0;
    auto take = [&](size_t bytes) -> char* {
        char* p = ws + off;
        off = (off + bytes + 255) & ~(size_t)255;
        return p;
    };
    int*    topi      = (int*)   take((size_t)T_ * 2 * sizeof(int));
    float*  topw      = (float*) take((size_t)T_ * 2 * sizeof(float));
    int*    blockcnt  = (int*)   take((size_t)NB_ * 16 * sizeof(int));
    int*    blockbase = (int*)   take((size_t)NB_ * 16 * sizeof(int));
    int*    slot      = (int*)   take((size_t)T_ * 2 * sizeof(int));
    __bf16* Xbuf      = (__bf16*)take((size_t)E_ * C_ * H_ * sizeof(__bf16));
    __bf16* Hbuf      = (__bf16*)take((size_t)E_ * C_ * F_ * sizeof(__bf16));
    float*  Ybuf      = (float*) take((size_t)E_ * C_ * H_ * sizeof(float));

    // 1) router
    moe_router<<<T_ / 8, 256, 0, stream>>>(tokens, w_router, topi, topw);
    // 2-4) deterministic slot assignment
    moe_hist<<<NB_, 256, 0, stream>>>(topi, blockcnt);
    moe_scan<<<1, 32, 0, stream>>>(blockcnt, blockbase);
    moe_rank<<<NB_, 256, 0, stream>>>(topi, blockbase, slot);
    // 5) zero Xbuf
    {
        size_t chunks16 = ((size_t)E_ * C_ * H_ * sizeof(__bf16)) / 16;
        moe_zero16<<<(unsigned)(chunks16 / 256), 256, 0, stream>>>((uint4*)Xbuf);
    }
    // 6) gather (one wave per (t,k) pair)
    moe_gather<<<(T_ * 2) / 8, 256, 0, stream>>>(tokens, topi, slot, Xbuf);
    // 7) fused gate/up GEMM + SiLU (double-buffered, async A staging)
    moe_gemm_gateup<<<dim3(F_ / 64, C_ / 128, E_), 256, 0, stream>>>(Xbuf, w_gate, w_up, Hbuf);
    // 8) down GEMM (double-buffered, async A staging)
    moe_gemm_down<<<dim3(H_ / 128, C_ / 128, E_), 256, 0, stream>>>(Hbuf, w_down, Ybuf);
    // 9) weighted scatter
    moe_scatter<<<T_ / 8, 256, 0, stream>>>(Ybuf, topi, slot, topw, out);
}
